// AttnBlock_25847113188100
// MI455X (gfx1250) — matrix operand order repaired and checked by reading, not compiled, not run on hardware
//
#include <hip/hip_runtime.h>
#include <cstdint>

// ---------------------------------------------------------------------------
// CDNA5 (gfx1250) fused attention block:
//   q = LN_head(x1 @ wq + bq), k = LN_head(kv[:,:C]), v = kv[:,C:]
//   out = softmax(q k^T / 8) v     (flash-attention, bf16 WMMA, fp32 accum)
// Fragments load as contiguous 16B runs (ds_load_b128); B-side operands are
// staged TRANSPOSED in LDS; fragment loads are batched ahead of WMMA groups
// so the ds loads clause together and WMMAs issue back-to-back.
// ---------------------------------------------------------------------------

typedef __bf16 bf16_t;
typedef __attribute__((ext_vector_type(16))) __bf16 v16bf;
typedef __attribute__((ext_vector_type(8)))  __bf16 v8bf;
typedef __attribute__((ext_vector_type(8)))  float  v8f;

#define DIM    1024
#define NHEADS 16
#define HDIM   64
#define BATCH  4
#define SEQ    2048

// 16-bit A 16x32 per-lane K runs: {8h..8h+7} and {16+8h..16+8h+7}
// 16-bit B 32x16 per-lane K runs: same as A; this file loads A and B as {16h..16h+15}   (h = lane>>4)

__device__ __forceinline__ v8f v8f_zero() { v8f z = {}; return z; }

__device__ __forceinline__ v16bf cat8(v8bf lo, v8bf hi) {
  v16bf r;
#pragma unroll
  for (int i = 0; i < 8; ++i) { r[i] = lo[i]; r[8 + i] = hi[i]; }
  return r;
}
__device__ __forceinline__ v8bf ld8(const bf16_t* p) {
  return *(const v8bf*)p;  // 16B-aligned by construction -> ds_load_b128
}

// ---------------------------------------------------------------------------
// GEMM (M x 1024) x (1024 x ncols) + bias, then per-head-64 LayerNorm (or raw
// copy for the V half of the KV projection). Output bf16 in (B*H, N, 64).
// Block tile: 64 rows x 128 cols, 8 waves, each wave 16x64 (4 WMMA accums).
// ---------------------------------------------------------------------------
__global__ __launch_bounds__(256)
void proj_gemm_ln_kernel(const float* __restrict__ X,
                         const float* __restrict__ W,
                         const float* __restrict__ bias,
                         const float* __restrict__ gamma,
                         const float* __restrict__ beta,
                         bf16_t* __restrict__ outLN,
                         bf16_t* __restrict__ outV,
                         int ncols, int vsplit) {
  __shared__ bf16_t As[64][32];        // row-major A tile
  __shared__ bf16_t BsT[128][40];      // TRANSPOSED B tile: [col][k], pad 40
  __shared__ float  Cs[64][128];

  const int tid  = threadIdx.x;
  const int lane = tid & 31;
  const int wave = tid >> 5;
  const int half = lane >> 4;
  const int mrow = lane & 15;
  const int wr   = wave & 3;   // row group (4 x 16 rows)
  const int wc   = wave >> 2;  // col group (2 x 64 cols)

  const int col_base = blockIdx.x * 128;
  const int row_base = blockIdx.y * 64;

  // Staging coordinates
  const int rA  = tid >> 2;            // 64 rows, 4 threads/row
  const int cA  = (tid & 3) * 8;       // 8 floats each
  const int rB  = tid >> 3;            // 32 k-rows, 8 threads/row
  const int cB  = (tid & 7) * 16;      // 16 cols each

  v8f acc[4];
#pragma unroll
  for (int t = 0; t < 4; ++t) acc[t] = v8f_zero();

  for (int k0 = 0; k0 < DIM; k0 += 32) {
    __syncthreads();
    if (k0 + 32 < DIM) {  // gfx1250 global_prefetch_b8 for the next K-slice
      __builtin_prefetch(&X[(size_t)(row_base + rA) * DIM + k0 + 32 + cA], 0, 1);
      __builtin_prefetch(&W[(size_t)(k0 + 32 + rB) * ncols + col_base + cB], 0, 1);
    }
    // Stage A tile 64x32 (f32 -> bf16), one ds_store_b128 per thread
    {
      const float* xp = &X[(size_t)(row_base + rA) * DIM + k0 + cA];
      v8bf a;
#pragma unroll
      for (int i = 0; i < 8; ++i) a[i] = (bf16_t)xp[i];
      *(v8bf*)&As[rA][cA] = a;
    }
    // Stage B tile 32x128 transposed: BsT[col][k] = W[k0+k][col_base+col]
    {
      const float* wp = &W[(size_t)(k0 + rB) * ncols + col_base + cB];
#pragma unroll
      for (int i = 0; i < 16; ++i) BsT[cB + i][rB] = (bf16_t)wp[i];
    }
    __syncthreads();

    // Batch all fragment loads, then issue the 4 WMMAs back-to-back.
    const bf16_t* ap = &As[wr * 16 + mrow][0];
    v16bf afrag = cat8(ld8(ap + 16 * half), ld8(ap + 16 * half + 8));
    v16bf bfrag[4];
#pragma unroll
    for (int t = 0; t < 4; ++t) {
      const bf16_t* bp = &BsT[wc * 64 + t * 16 + mrow][16 * half];
      bfrag[t] = cat8(ld8(bp), ld8(bp + 8));
    }
#pragma unroll
    for (int t = 0; t < 4; ++t)
      acc[t] = __builtin_amdgcn_wmma_f32_16x16x32_bf16(
          false, afrag, false, bfrag[t], (short)0, acc[t], false, false);
  }

  __syncthreads();
  // Accumulators + bias -> LDS f32 tile (C layout: VGPR i = row i / i+8)
#pragma unroll
  for (int t = 0; t < 4; ++t) {
    int col  = wc * 64 + t * 16 + mrow;
    float bv = bias[col_base + col];
#pragma unroll
    for (int i = 0; i < 8; ++i) {
      int row = half ? (i + 8) : i;
      Cs[wr * 16 + row][col] = acc[t][i] + bv;
    }
  }
  __syncthreads();

  const bool isV = (vsplit != 0) && (col_base >= vsplit);
  if (!isV) {
    // Per-row LayerNorm over the 64-wide head segment; 128 segments / block.
    if (tid < 128) {
      int r  = tid & 63;
      int hs = tid >> 6;
      const float* rp = &Cs[r][hs * 64];
      float s = 0.f;
#pragma unroll 8
      for (int d = 0; d < 64; ++d) s += rp[d];
      float mean = s * (1.f / 64.f);
      float var = 0.f;
#pragma unroll 8
      for (int d = 0; d < 64; ++d) { float df = rp[d] - mean; var += df * df; }
      float rstd = rsqrtf(var * (1.f / 64.f) + 1e-5f);
      int cglob = col_base + hs * 64;
      int h = cglob >> 6;           // head index (cglob < 1024 here)
      int m = row_base + r;
      int b = m >> 11;
      int n = m & (SEQ - 1);
      bf16_t* o = outLN + (((size_t)(b * NHEADS + h)) * SEQ + n) * HDIM;
#pragma unroll 8
      for (int d = 0; d < 64; ++d) {
        float val = (rp[d] - mean) * rstd * gamma[d] + beta[d];
        o[d] = (bf16_t)val;
      }
    }
  } else {
    // V half of KV projection: bias already applied, just store bf16.
    for (int l = tid; l < 64 * 128; l += 256) {
      int r = l >> 7, c = l & 127;
      int cglob = col_base - vsplit + c;
      int h = cglob >> 6, d = cglob & 63;
      int m = row_base + r;
      int b = m >> 11, n = m & (SEQ - 1);
      outV[(((size_t)(b * NHEADS + h)) * SEQ + n) * HDIM + d] = (bf16_t)Cs[r][c];
    }
  }
}

// ---------------------------------------------------------------------------
// Flash attention: per (b,h) head, 128 q-rows per block (16 per wave),
// K/V tiled 64 rows at a time. K tile staged with gfx1250 async global->LDS
// (ASYNCcnt), V tile staged transposed for contiguous B-fragment loads.
// ---------------------------------------------------------------------------
__global__ __launch_bounds__(256)
void flash_attn_kernel(const bf16_t* __restrict__ Q,
                       const bf16_t* __restrict__ K,
                       const bf16_t* __restrict__ V,
                       float* __restrict__ Out) {
  __shared__ bf16_t Ks[64][64];        // row-major K tile (async-loaded)
  __shared__ bf16_t VsT[64][72];       // TRANSPOSED V tile: [d][j], pad 72
  __shared__ bf16_t Ps[8][16][64];     // per-wave P tile (C->A layout bounce)

  const int tid  = threadIdx.x;
  const int lane = tid & 31;
  const int wave = tid >> 5;
  const int half = lane >> 4;
  const int mrow = lane & 15;

  const int bh  = blockIdx.y;                    // b*16 + h
  const int qr0 = blockIdx.x * 128 + wave * 16;  // this wave's first q row

  // Q fragments for rows qr0..qr0+15, d = 0..63 (two K=32 frags), from global.
  v16bf qa[2];
  {
    const bf16_t* qp = Q + ((size_t)bh * SEQ + qr0 + mrow) * HDIM;
#pragma unroll
    for (int f = 0; f < 2; ++f)
      qa[f] = cat8(ld8(qp + f * 32 + 16 * half),
                   ld8(qp + f * 32 + 16 * half + 8));
  }

  float Mi[8], Li[8];
#pragma unroll
  for (int i = 0; i < 8; ++i) { Mi[i] = -1e30f; Li[i] = 0.f; }
  v8f acc[4];
#pragma unroll
  for (int t = 0; t < 4; ++t) acc[t] = v8f_zero();

  const float scale = 0.125f;  // 1/sqrt(64)

  const bf16_t* kgb = K + (size_t)bh * SEQ * HDIM;
  const bf16_t* vgb = V + (size_t)bh * SEQ * HDIM;
  const uint32_t ks_lds = (uint32_t)(uintptr_t)&Ks[0][0];  // LDS byte offset

  // V staging coords: 64x64 tile, 16 shorts per thread, transposed store
  const int vj = tid >> 2;             // source row j (64 rows, 4 thr/row)
  const int vd = (tid & 3) * 16;       // source col d

  for (int kb = 0; kb < SEQ; kb += 64) {
    __syncthreads();
    // ---- K tile: async global->LDS b128, 2 transfers per thread ----
#pragma unroll
    for (int i = 0; i < 2; ++i) {
      int l = tid + i * 256;           // 512 x 16B = 8KB
      uint64_t ga = (uint64_t)(uintptr_t)(kgb + (size_t)kb * HDIM) + l * 16;
      uint32_t la = ks_lds + l * 16;
      asm volatile("global_load_async_to_lds_b128 %0, %1, off"
                   :: "v"(la), "v"(ga) : "memory");
    }
    // ---- V tile: read 16 contiguous shorts, scatter transposed ----
    {
      const bf16_t* vp = vgb + (size_t)(kb + vj) * HDIM + vd;
      v8bf x0 = ld8(vp), x1 = ld8(vp + 8);
#pragma unroll
      for (int i = 0; i < 8; ++i) {
        VsT[vd + i][vj]     = x0[i];
        VsT[vd + 8 + i][vj] = x1[i];
      }
    }
    asm volatile("s_wait_asynccnt 0x0" ::: "memory");
    __syncthreads();

    // ---- scores = q @ k^T: per K-phase, batch 4 B-frag loads then 4 WMMAs
    v8f s[4];
#pragma unroll
    for (int t = 0; t < 4; ++t) s[t] = v8f_zero();
#pragma unroll
    for (int f = 0; f < 2; ++f) {
      v16bf kf[4];
#pragma unroll
      for (int t = 0; t < 4; ++t) {
        const bf16_t* kr = &Ks[t * 16 + mrow][f * 32 + 16 * half];
        kf[t] = cat8(ld8(kr), ld8(kr + 8));
      }
#pragma unroll
      for (int t = 0; t < 4; ++t)
        s[t] = __builtin_amdgcn_wmma_f32_16x16x32_bf16(
            false, qa[f], false, kf[t], (short)0, s[t], false, false);
    }

    // ---- online softmax (xor-shuffles 1..8 stay inside 16-lane group) ----
    float alpha[8], rsum[8];
#pragma unroll
    for (int i = 0; i < 8; ++i) {
      float m = -1e30f;
#pragma unroll
      for (int t = 0; t < 4; ++t) {
        float sv = s[t][i] * scale;
        s[t][i]  = sv;
        m = fmaxf(m, sv);
      }
#pragma unroll
      for (int off = 1; off < 16; off <<= 1)
        m = fmaxf(m, __shfl_xor(m, off, 32));
      float mnew = fmaxf(Mi[i], m);
      alpha[i]   = __expf(Mi[i] - mnew);
      Mi[i]      = mnew;
      float psum = 0.f;
      int prow   = half ? (i + 8) : i;
#pragma unroll
      for (int t = 0; t < 4; ++t) {
        float p = __expf(s[t][i] - mnew);
        psum += p;
        Ps[wave][prow][t * 16 + mrow] = (bf16_t)p;
      }
#pragma unroll
      for (int off = 1; off < 16; off <<= 1)
        psum += __shfl_xor(psum, off, 32);
      rsum[i] = psum;
    }
#pragma unroll
    for (int t = 0; t < 4; ++t)
#pragma unroll
      for (int i = 0; i < 8; ++i) acc[t][i] *= alpha[i];
#pragma unroll
    for (int i = 0; i < 8; ++i) Li[i] = Li[i] * alpha[i] + rsum[i];

    // ---- out += P @ V: per K-phase, batch A-frag + 4 B-frag loads ----
    const bf16_t* pp = &Ps[wave][mrow][0];
#pragma unroll
    for (int f = 0; f < 2; ++f) {
      v16bf pa = cat8(ld8(pp + f * 32 + 16 * half),
                      ld8(pp + f * 32 + 16 * half + 8));
      v16bf vf[4];
#pragma unroll
      for (int t = 0; t < 4; ++t) {
        const bf16_t* vr = &VsT[t * 16 + mrow][f * 32 + 16 * half];
        vf[t] = cat8(ld8(vr), ld8(vr + 8));
      }
#pragma unroll
      for (int t = 0; t < 4; ++t)
        acc[t] = __builtin_amdgcn_wmma_f32_16x16x32_bf16(
            false, pa, false, vf[t], (short)0, acc[t], false, false);
    }
  }

  // ---- epilogue: normalize by running denominator, store fp32 (B,N,C) ----
  const int b = bh >> 4, h = bh & 15;
#pragma unroll
  for (int i = 0; i < 8; ++i) {
    float inv = 1.0f / Li[i];
    int row   = half ? (i + 8) : i;
    int n     = qr0 + row;
#pragma unroll
    for (int t = 0; t < 4; ++t) {
      int d = t * 16 + mrow;
      Out[((size_t)b * SEQ + n) * DIM + h * HDIM + d] = acc[t][i] * inv;
    }
  }
}

// ---------------------------------------------------------------------------
extern "C" void kernel_launch(void* const* d_in, const int* in_sizes, int n_in,
                              void* d_out, int out_size, void* d_ws, size_t ws_size,
                              hipStream_t stream) {
  const float* x1   = (const float*)d_in[0];
  const float* x2   = (const float*)d_in[1];
  const float* wq   = (const float*)d_in[2];
  const float* bq   = (const float*)d_in[3];
  const float* wkv  = (const float*)d_in[4];
  const float* bkv  = (const float*)d_in[5];
  const float* gq   = (const float*)d_in[6];
  const float* btq  = (const float*)d_in[7];
  const float* gk   = (const float*)d_in[8];
  const float* btk  = (const float*)d_in[9];
  float* out = (float*)d_out;

  const size_t headElems = (size_t)BATCH * NHEADS * SEQ * HDIM;  // 8,388,608
  bf16_t* qws = (bf16_t*)d_ws;
  bf16_t* kws = qws + headElems;
  bf16_t* vws = kws + headElems;

  dim3 blk(256);
  // Q projection + head LayerNorm  (M=8192, ncols=1024)
  proj_gemm_ln_kernel<<<dim3(1024 / 128, (BATCH * SEQ) / 64), blk, 0, stream>>>(
      x1, wq, bq, gq, btq, qws, nullptr, 1024, 0);
  // KV projection: K half gets LayerNorm, V half stored raw (ncols=2048)
  proj_gemm_ln_kernel<<<dim3(2048 / 128, (BATCH * SEQ) / 64), blk, 0, stream>>>(
      x2, wkv, bkv, gk, btk, kws, vws, 2048, 1024);
  // Flash attention over 64 (b,h) heads, 16 q-tiles of 128 rows each
  flash_attn_kernel<<<dim3(SEQ / 128, BATCH * NHEADS), blk, 0, stream>>>(
      qws, kws, vws, out);
}
